// PerlinAttention_80539226734941
// MI455X (gfx1250) — matrix operand order repaired and checked by reading, not compiled, not run on hardware
//
#include <hip/hip_runtime.h>
#include <hip/hip_bf16.h>
#include <math.h>

// ---------------------------------------------------------------------------
// PerlinAttention pipeline for MI455X (gfx1250, wave32, WMMA bf16 + TDM)
//   B=1, H=12, T=2048, D=64, M=266 (pad 272), L=128
// ---------------------------------------------------------------------------

typedef __attribute__((ext_vector_type(16))) __bf16 v16bf;
typedef __attribute__((ext_vector_type(8)))  float  v8f;
typedef __attribute__((ext_vector_type(4)))  unsigned int v4u;
typedef __attribute__((ext_vector_type(8)))  int    v8i;
typedef __attribute__((ext_vector_type(4)))  int    v4i;

#define H_    12
#define T_    2048
#define D_    64
#define M_    266
#define MP_   272
#define L_    128
#define ROWS_ 24576   // B*H*T

// CDNA5 16-bit A-matrix 16x32 layout: packed element i, lane-half hsel -> K
__device__ __forceinline__ int amapK(int i, int hsel) {
  return (i < 8 ? i : i + 8) + hsel * 8;
}

__device__ __forceinline__ void pack8(v16bf& a, int base, float4 f0, float4 f1,
                                      float scale) {
  a[base + 0] = (__bf16)(f0.x * scale);
  a[base + 1] = (__bf16)(f0.y * scale);
  a[base + 2] = (__bf16)(f0.z * scale);
  a[base + 3] = (__bf16)(f0.w * scale);
  a[base + 4] = (__bf16)(f1.x * scale);
  a[base + 5] = (__bf16)(f1.y * scale);
  a[base + 6] = (__bf16)(f1.z * scale);
  a[base + 7] = (__bf16)(f1.w * scale);
}

__device__ __forceinline__ unsigned ldsOffset(const void* p) {
  // LDS aperture generic addresses carry the workgroup-relative byte offset
  // in the low 32 bits (CDNA5 ISA 10.2 aperture mapping).
  return (unsigned)(unsigned long long)(uintptr_t)p;
}

// ---------------------------------------------------------------------------
// Tensor Data Mover: stage `nelem` fp32 (contiguous) from global into LDS.
// D# per CDNA5 ISA ch.8: group0 = {count=1, lds_addr, global_addr, type=2},
// group1 = {data_size=4B, 1-D tile: tensor_dim0=tile_dim0=nelem, dim1=1}.
// ---------------------------------------------------------------------------
__device__ __forceinline__ void tdm_load_1d(const void* gptr, unsigned lds_off,
                                            unsigned nelem) {
  unsigned long long ga = (unsigned long long)(uintptr_t)gptr;
  v4u g0;
  g0[0] = 1u;                                        // count=1, user-mode D#
  g0[1] = lds_off;                                   // lds_addr (bytes)
  g0[2] = (unsigned)(ga & 0xFFFFFFFFull);            // global_addr[31:0]
  g0[3] = (unsigned)((ga >> 32) & 0x01FFFFFFull) | (2u << 30);  // addr[56:32], type=2
  v8i g1;
  g1[0] = (int)(2u << 16);                           // data_size=2 (4 bytes)
  g1[1] = (int)((nelem & 0xFFFFu) << 16);            // tensor_dim0[15:0] @ bits 63:48
  g1[2] = (int)((nelem >> 16) | (1u << 16));         // tensor_dim0[31:16], tensor_dim1=1
  g1[3] = (int)((nelem & 0xFFFFu) << 16);            // tile_dim0 (nelem < 65536)
  g1[4] = 1;                                         // tile_dim1=1, tile_dim2=0
  g1[5] = (int)nelem;                                // tensor_dim0_stride[31:0]
  g1[6] = 0;
  g1[7] = 0;
  v4i gz = {0, 0, 0, 0};
#if defined(__clang_major__) && (__clang_major__ >= 23)
  v8i gz8 = {0, 0, 0, 0, 0, 0, 0, 0};
  __builtin_amdgcn_tensor_load_to_lds(g0, g1, gz, gz, gz8, 0);
#else
  __builtin_amdgcn_tensor_load_to_lds(g0, g1, gz, gz, 0);
#endif
}

// ---------------------------------------------------------------------------
// 1) FAVOR projection GEMM: U = (dn*X) @ proj^T   [24576x64]x[64x266]
// ---------------------------------------------------------------------------
__global__ __launch_bounds__(256)
void favor_u_gemm(const float* __restrict__ x, const float* __restrict__ proj,
                  float* __restrict__ u) {
  const float dn = 0.35355339059327373f;  // 64^-0.25
  int wid = threadIdx.x >> 5, lane = threadIdx.x & 31;
  int tile = blockIdx.x * 8 + wid;
  if (tile >= 1536 * 17) return;
  int rt = tile / 17, mt = tile % 17;
  int row0 = rt * 16, m0 = mt * 16;
  int hsel = lane >> 4, l16 = lane & 15;
  int arow = row0 + l16;
  int bn = m0 + l16;
  int bnc = bn < M_ ? bn : M_ - 1;          // clamp: unconditional load + select
  v8f acc = {};
#pragma unroll
  for (int kk = 0; kk < 64; kk += 32) {
    v16bf a, b;
    const float4* ap = (const float4*)(x + arow * 64 + kk + hsel * 8);
    pack8(a, 0, ap[0], ap[1], dn);
    pack8(a, 8, ap[4], ap[5], dn);
#pragma unroll
    for (int i = 0; i < 16; ++i) {
      float bv = proj[bnc * 64 + kk + amapK(i, hsel)];
      b[i] = (__bf16)(bn < M_ ? bv : 0.0f);
    }
    acc = __builtin_amdgcn_wmma_f32_16x16x32_bf16(false, a, false, b,
                                                  (short)0, acc, false, false);
  }
#pragma unroll
  for (int j = 0; j < 8; ++j)
    u[(row0 + j + hsel * 8) * MP_ + (m0 + l16)] = acc[j];
}

// ---------------------------------------------------------------------------
// 2) Global max over ku (key stabilizer), two stage
// ---------------------------------------------------------------------------
__global__ __launch_bounds__(256)
void ku_rowmax_partial(const float* __restrict__ ku, float* __restrict__ part) {
  __shared__ float red[256];
  int row = blockIdx.x * 256 + threadIdx.x;
  float m = -3.4e38f;
  if (row < ROWS_)
    for (int j = 0; j < M_; ++j) m = fmaxf(m, ku[row * MP_ + j]);
  red[threadIdx.x] = m; __syncthreads();
  for (int s = 128; s > 0; s >>= 1) {
    if (threadIdx.x < s) red[threadIdx.x] = fmaxf(red[threadIdx.x], red[threadIdx.x + s]);
    __syncthreads();
  }
  if (threadIdx.x == 0) part[blockIdx.x] = red[0];
}

__global__ __launch_bounds__(128)
void ku_max_final(const float* __restrict__ part, int n, float* __restrict__ out) {
  __shared__ float red[128];
  float m = -3.4e38f;
  for (int i = threadIdx.x; i < n; i += 128) m = fmaxf(m, part[i]);
  red[threadIdx.x] = m; __syncthreads();
  for (int s = 64; s > 0; s >>= 1) {
    if (threadIdx.x < s) red[threadIdx.x] = fmaxf(red[threadIdx.x], red[threadIdx.x + s]);
    __syncthreads();
  }
  if (threadIdx.x == 0) out[0] = red[0];
}

// ---------------------------------------------------------------------------
// 3) FAVOR exp: f = ratio*(exp(u - diag - stab) + 1e-4), in place over u.
// ---------------------------------------------------------------------------
__global__ __launch_bounds__(256)
void favor_exp(float* __restrict__ uf, const float* __restrict__ x,
               const float* __restrict__ stabp, int isQuery) {
  int wid = threadIdx.x >> 5, lane = threadIdx.x & 31;
  int row = blockIdx.x * 8 + wid;
  if (row >= ROWS_) return;
  float x0 = x[row * 64 + lane], x1 = x[row * 64 + lane + 32];
  float ss = x0 * x0 + x1 * x1;
  for (int o = 16; o > 0; o >>= 1) ss += __shfl_xor(ss, o, 32);
  float diag = 0.5f * 0.125f * ss;  // dn^2 = 64^-0.5 = 0.125
  float stab;
  if (isQuery) {
    float m = -3.4e38f;
    for (int j = lane; j < M_; j += 32) m = fmaxf(m, uf[row * MP_ + j]);
    for (int o = 16; o > 0; o >>= 1) m = fmaxf(m, __shfl_xor(m, o, 32));
    stab = m;
  } else {
    stab = stabp[0];
  }
  const float ratio = 0.06131393394849658f;  // 266^-0.5
  for (int j = lane; j < MP_; j += 32) {
    float v = 0.f;
    if (j < M_) v = ratio * (__expf(uf[row * MP_ + j] - diag - stab) + 1e-4f);
    uf[row * MP_ + j] = v;
  }
}

// ---------------------------------------------------------------------------
// 4) Causal linear attention scan. One workgroup per head; 266x64 running
//    context lives in LDS (~71 KB of the 320 KB/WGP).
// ---------------------------------------------------------------------------
__global__ __launch_bounds__(256)
void causal_attn(const float* __restrict__ qf, const float* __restrict__ kf,
                 const float* __restrict__ v, float* __restrict__ perf) {
  extern __shared__ float sm[];
  float* ccs  = sm;               // M_*64
  float* kcs  = ccs + M_ * 64;    // MP_
  float* qrow = kcs + MP_;        // MP_
  float* krow = qrow + MP_;       // MP_
  float* vrow = krow + MP_;       // 64
  float* red  = vrow + 64;        // 256
  int tid = threadIdx.x, h = blockIdx.x;
  for (int i = tid; i < M_ * 64; i += 256) ccs[i] = 0.f;
  for (int i = tid; i < MP_; i += 256) kcs[i] = 0.f;
  __syncthreads();
  int d = tid & 63, mg = tid >> 6;  // thread owns (m = mg mod 4, d) slice
  for (int t = 0; t < T_; ++t) {
    long rowf = (long)h * T_ + t;
    for (int i = tid; i < MP_; i += 256) {
      qrow[i] = qf[rowf * MP_ + i];
      krow[i] = kf[rowf * MP_ + i];
    }
    if (tid < 64) vrow[tid] = v[rowf * 64 + tid];
    __syncthreads();
    for (int i = tid; i < MP_; i += 256) kcs[i] += krow[i];  // inclusive cumsum
    __syncthreads();
    float p = 0.f;
    for (int m = tid; m < M_; m += 256) p += qrow[m] * (kcs[m] + 1e-6f);
    red[tid] = p; __syncthreads();
    if (tid < 64) red[tid] += red[tid + 64] + red[tid + 128] + red[tid + 192];
    __syncthreads();
    if (tid == 0) {
      float s = 0.f;
      for (int i = 0; i < 64; ++i) s += red[i];
      red[0] = 1.0f / s;
    }
    __syncthreads();
    float dinv = red[0];
    float outv = 0.f, vd = vrow[d];
    for (int m = mg; m < M_; m += 4) {           // disjoint LDS ownership
      float c = ccs[m * 64 + d] + krow[m] * vd;  // ctx cumsum
      ccs[m * 64 + d] = c;
      outv += qrow[m] * c;
    }
    __syncthreads();
    red[tid] = outv; __syncthreads();
    if (tid < 64)
      perf[rowf * 64 + d] = (red[d] + red[d + 64] + red[d + 128] + red[d + 192]) * dinv;
    __syncthreads();
  }
}

// ---------------------------------------------------------------------------
// 5) Encoder: h1 = gelu(LN(pv @ w_enc + b_enc)); w_enc staged in LDS by TDM.
//    pv segment is chosen by a compile-time-unrolled uniform branch on kk.
// ---------------------------------------------------------------------------
__global__ __launch_bounds__(256)
void encoder_gemm(const float* __restrict__ perf, const float* __restrict__ v,
                  const float* __restrict__ wenc, const float* __restrict__ benc,
                  const float* __restrict__ lng, const float* __restrict__ lnb,
                  float* __restrict__ h1) {
  extern __shared__ float esm[];
  float* lw    = esm;                 // 192*128 staged weights
  float* shbuf = esm + 192 * 128;     // 16*132 activation tile
  float* mu    = shbuf + 16 * 132;    // 16
  float* rs    = mu + 16;             // 16
  if (threadIdx.x < 32) {
    tdm_load_1d(wenc, ldsOffset(lw), 192 * 128);
    __builtin_amdgcn_s_wait_tensorcnt(0);
  }
  __syncthreads();
  int wid = threadIdx.x >> 5, lane = threadIdx.x & 31;
  int hsel = lane >> 4, l16 = lane & 15;
  int row0 = blockIdx.x * 16, n0 = wid * 16;
  int arow = row0 + l16;
  const float* pb = perf + arow * 64;
  const float* vb = v + arow * 64;
  v8f acc = {};
#pragma unroll
  for (int kk = 0; kk < 192; kk += 32) {
    int seg = kk >> 6;                 // 0: perf | 1: v | 2: perf*v  (uniform)
    int cb = (kk & 63) + hsel * 8;     // both 8-runs stay inside one segment
    const float4* pp = (const float4*)(pb + cb);
    const float4* vv = (const float4*)(vb + cb);
    float4 f0, f1, f2, f3;
    if (seg == 0)      { f0 = pp[0]; f1 = pp[1]; f2 = pp[4]; f3 = pp[5]; }
    else if (seg == 1) { f0 = vv[0]; f1 = vv[1]; f2 = vv[4]; f3 = vv[5]; }
    else {
      float4 a0 = pp[0], a1 = pp[1], a2 = pp[4], a3 = pp[5];
      float4 b0 = vv[0], b1 = vv[1], b2 = vv[4], b3 = vv[5];
      f0 = make_float4(a0.x*b0.x, a0.y*b0.y, a0.z*b0.z, a0.w*b0.w);
      f1 = make_float4(a1.x*b1.x, a1.y*b1.y, a1.z*b1.z, a1.w*b1.w);
      f2 = make_float4(a2.x*b2.x, a2.y*b2.y, a2.z*b2.z, a2.w*b2.w);
      f3 = make_float4(a3.x*b3.x, a3.y*b3.y, a3.z*b3.z, a3.w*b3.w);
    }
    v16bf a, b;
    pack8(a, 0, f0, f1, 1.0f);
    pack8(a, 8, f2, f3, 1.0f);
#pragma unroll
    for (int i = 0; i < 16; ++i)
      b[i] = (__bf16)lw[(kk + amapK(i, hsel)) * 128 + n0 + l16];
    acc = __builtin_amdgcn_wmma_f32_16x16x32_bf16(false, a, false, b,
                                                  (short)0, acc, false, false);
  }
#pragma unroll
  for (int j = 0; j < 8; ++j)
    shbuf[(j + hsel * 8) * 132 + n0 + l16] = acc[j] + benc[n0 + l16];
  __syncthreads();
  if (threadIdx.x < 16) {
    int r = threadIdx.x;
    float s = 0.f, s2 = 0.f;
    for (int c = 0; c < 128; ++c) { float xv = shbuf[r * 132 + c]; s += xv; s2 += xv * xv; }
    float m = s * (1.0f / 128.0f);
    mu[r] = m; rs[r] = rsqrtf(s2 * (1.0f / 128.0f) - m * m + 1e-5f);
  }
  __syncthreads();
  for (int idx = threadIdx.x; idx < 16 * 128; idx += 256) {
    int r = idx >> 7, c = idx & 127;
    float xv = (shbuf[r * 132 + c] - mu[r]) * rs[r] * lng[c] + lnb[c];
    h1[(row0 + r) * 128 + c] = 0.5f * xv * (1.0f + erff(xv * 0.70710678f));
  }
}

// ---------------------------------------------------------------------------
// 6) Decoder: dec = h1 @ w_dec + b_dec; w_dec (64 KB) staged in LDS by TDM.
// ---------------------------------------------------------------------------
__global__ __launch_bounds__(256)
void decoder_gemm(const float* __restrict__ h1, const float* __restrict__ wdec,
                  const float* __restrict__ bdec, float* __restrict__ dec) {
  extern __shared__ float lw[];       // 128*128
  if (threadIdx.x < 32) {
    tdm_load_1d(wdec, ldsOffset(lw), 128 * 128);
    __builtin_amdgcn_s_wait_tensorcnt(0);
  }
  __syncthreads();
  int wid = threadIdx.x >> 5, lane = threadIdx.x & 31;
  int hsel = lane >> 4, l16 = lane & 15;
  int row0 = blockIdx.x * 16, n0 = wid * 16;
  int arow = row0 + l16;
  v8f acc = {};
#pragma unroll
  for (int kk = 0; kk < 128; kk += 32) {
    v16bf a, b;
    const float4* ap = (const float4*)(h1 + arow * 128 + kk + hsel * 8);
    pack8(a, 0, ap[0], ap[1], 1.0f);
    pack8(a, 8, ap[4], ap[5], 1.0f);
#pragma unroll
    for (int i = 0; i < 16; ++i)
      b[i] = (__bf16)lw[(kk + amapK(i, hsel)) * 128 + n0 + l16];
    acc = __builtin_amdgcn_wmma_f32_16x16x32_bf16(false, a, false, b,
                                                  (short)0, acc, false, false);
  }
#pragma unroll
  for (int j = 0; j < 8; ++j)
    dec[(row0 + j + hsel * 8) * 128 + n0 + l16] = acc[j] + bdec[n0 + l16];
}

// ---------------------------------------------------------------------------
// 7) Generic causal 3x3 conv as implicit GEMM; filters staged in LDS by TDM.
//    Pad=((2,0),(1,1)); optional stride 2, nearest-2x upsampled input,
//    residual add, relu. Bounds handled by clamp + select (no EXEC branches).
// ---------------------------------------------------------------------------
__global__ __launch_bounds__(256)
void conv3x3_wmma(const float* __restrict__ in, const float* __restrict__ w,
                  const float* __restrict__ bias, const float* __restrict__ res,
                  float* __restrict__ out,
                  int Cin, int Cout, int Hin, int Win,
                  int Hout, int wshift, int sh, int sw, int ups, int doRelu,
                  int ntN, int ntTot) {
  extern __shared__ float lw[];       // Cout*Cin*9 filters
  int K = Cin * 9;
  if (threadIdx.x < 32) {
    tdm_load_1d(w, ldsOffset(lw), (unsigned)(Cout * K));
    __builtin_amdgcn_s_wait_tensorcnt(0);
  }
  __syncthreads();
  int wid = threadIdx.x >> 5, lane = threadIdx.x & 31;
  int tile = blockIdx.x * 8 + wid;
  if (tile < ntTot) {                 // wave-uniform guard (EXEC stays full)
    int nt = tile % ntN, rt = tile / ntN;
    int s0 = rt * 16, n0 = nt * 16;
    int hsel = lane >> 4, l16 = lane & 15;
    int Wout = 1 << wshift;
    int s = s0 + l16;
    int y = s >> wshift, x = s & (Wout - 1);
    int Hc = Hin << ups, Wc = Win << ups;
    int HinWin = Hin * Win;
    int bn = n0 + l16;
    int bnc = bn < Cout ? bn : Cout - 1;
    v8f acc = {};
    for (int kk = 0; kk < K; kk += 32) {
      __builtin_prefetch(in + (y * sh) * Win, 0, 2);   // global_prefetch_b8
      v16bf a, b;
#pragma unroll
      for (int i = 0; i < 16; ++i) {
        int k = kk + amapK(i, hsel);
        int kc = k < K ? k : K - 1;
        int ci = kc / 9, rr = kc - ci * 9;
        int kh = rr / 3, kw = rr - kh * 3;
        int yi = y * sh + kh - 2;      // causal pad (2,0) in time
        int xi = x * sw + kw - 1;      // pad (1,1) in width
        bool inb = (k < K) & (yi >= 0) & (yi < Hc) & (xi >= 0) & (xi < Wc);
        int yic = yi < 0 ? 0 : (yi >= Hc ? Hc - 1 : yi);
        int xic = xi < 0 ? 0 : (xi >= Wc ? Wc - 1 : xi);
        float av = in[ci * HinWin + (yic >> ups) * Win + (xic >> ups)];
        a[i] = (__bf16)(inb ? av : 0.0f);
        int bk = kk + amapK(i, hsel);
        int bkc = bk < K ? bk : K - 1;
        float bv = lw[bnc * K + bkc];
        b[i] = (__bf16)(((bk < K) & (bn < Cout)) ? bv : 0.0f);
      }
      acc = __builtin_amdgcn_wmma_f32_16x16x32_bf16(false, a, false, b,
                                                    (short)0, acc, false, false);
    }
    int HW = Hout << wshift;
#pragma unroll
    for (int j = 0; j < 8; ++j) {
      int so = s0 + j + hsel * 8;
      int co = n0 + l16;
      if (co < Cout) {
        float vv = acc[j] + bias[co];
        if (res) vv += res[co * HW + so];
        if (doRelu) vv = fmaxf(vv, 0.f);
        out[co * HW + so] = vv;
      }
    }
  }
}

// ---------------------------------------------------------------------------
// 8) scores = dec + cnn; softmax over L=128. One block per row.
// ---------------------------------------------------------------------------
__global__ __launch_bounds__(128)
void score_softmax(const float* __restrict__ dec, const float* __restrict__ c2o,
                   float* __restrict__ probs) {
  __shared__ float red[128];
  int r = blockIdx.x, tid = threadIdx.x;
  float val = dec[r * 128 + tid] + c2o[r * 128 + tid];
  red[tid] = val; __syncthreads();
  for (int s = 64; s > 0; s >>= 1) {
    if (tid < s) red[tid] = fmaxf(red[tid], red[tid + s]);
    __syncthreads();
  }
  float mx = red[0]; __syncthreads();
  float e = __expf(val - mx);
  red[tid] = e; __syncthreads();
  for (int s = 64; s > 0; s >>= 1) {
    if (tid < s) red[tid] += red[tid + s];
    __syncthreads();
  }
  probs[r * 128 + tid] = e / red[0];
}

// ---------------------------------------------------------------------------
// 9) Time-prefix sums of v per (h,d): prefV[h][t][d] = sum_{j<t} v[h][j][d]
// ---------------------------------------------------------------------------
__global__ __launch_bounds__(256)
void prefix_v(const float* __restrict__ v, float* __restrict__ pref) {
  int gid = blockIdx.x * 256 + threadIdx.x;
  if (gid >= H_ * 64) return;
  int h = gid >> 6, d = gid & 63;
  const float* vp = v + (size_t)h * T_ * 64 + d;
  float* pp = pref + (size_t)h * (T_ + 1) * 64 + d;
  float acc = 0.f;
  pp[0] = 0.f;
  for (int t = 0; t < T_; ++t) {
    acc += vp[(size_t)t * 64];
    pp[(size_t)(t + 1) * 64] = acc;
  }
}

// ---------------------------------------------------------------------------
// 10) Grid-sampled attention via contiguous buckets:
//     out[t,d] = (1/Z) sum_l probs[l]*(prefV[hi_l]-prefV[lo_l]),
//     lo_l = ceil((l*(t+1)-64)/128) clamped to [0,t+1], Z = sum probs*count.
// ---------------------------------------------------------------------------
__global__ __launch_bounds__(128)
void sparse_interp_out(const float* __restrict__ probs, const float* __restrict__ pref,
                       float* __restrict__ out) {
  __shared__ float sp[128];
  __shared__ int lo[129];
  __shared__ float red[128];
  int r = blockIdx.x;
  int h = r >> 11, t = r & (T_ - 1);
  int tid = threadIdx.x;
  sp[tid] = probs[r * 128 + tid];
  int tp1 = t + 1;
  int lov = (tid * tp1 + 63) >> 7;
  lo[tid] = (tid == 0) ? 0 : (lov < tp1 ? lov : tp1);
  if (tid == 0) lo[128] = tp1;
  __syncthreads();
  red[tid] = sp[tid] * (float)(lo[tid + 1] - lo[tid]);
  __syncthreads();
  for (int s = 64; s > 0; s >>= 1) {
    if (tid < s) red[tid] += red[tid + s];
    __syncthreads();
  }
  float zinv = 1.0f / (red[0] + 1e-6f);
  __syncthreads();
  int d = tid & 63, hsel = tid >> 6;
  const float* pb = pref + (size_t)h * (T_ + 1) * 64 + d;
  float acc = 0.f;
  for (int l = hsel * 64; l < hsel * 64 + 64; ++l)
    acc += sp[l] * (pb[(size_t)lo[l + 1] * 64] - pb[(size_t)lo[l] * 64]);
  red[tid] = acc; __syncthreads();
  if (tid < 64)
    out[(size_t)r * 64 + d] = (red[d] + red[d + 64]) * zinv;
}

// ---------------------------------------------------------------------------
extern "C" void kernel_launch(void* const* d_in, const int* in_sizes, int n_in,
                              void* d_out, int out_size, void* d_ws, size_t ws_size,
                              hipStream_t stream) {
  const float* q     = (const float*)d_in[0];
  const float* k     = (const float*)d_in[1];
  const float* v     = (const float*)d_in[2];
  const float* proj  = (const float*)d_in[3];
  const float* w_enc = (const float*)d_in[4];
  const float* b_enc = (const float*)d_in[5];
  const float* ln_g  = (const float*)d_in[6];
  const float* ln_b  = (const float*)d_in[7];
  const float* w_dec = (const float*)d_in[8];
  const float* b_dec = (const float*)d_in[9];
  const float* c1_w  = (const float*)d_in[10];
  const float* c1_b  = (const float*)d_in[11];
  const float* r1w1  = (const float*)d_in[12];
  const float* r1b1  = (const float*)d_in[13];
  const float* r1w2  = (const float*)d_in[14];
  const float* r1b2  = (const float*)d_in[15];
  const float* r2w1  = (const float*)d_in[16];
  const float* r2b1  = (const float*)d_in[17];
  const float* r2w2  = (const float*)d_in[18];
  const float* r2b2  = (const float*)d_in[19];
  const float* c2_w  = (const float*)d_in[20];
  const float* c2_b  = (const float*)d_in[21];
  float* out = (float*)d_out;
  float* ws  = (float*)d_ws;

  size_t off = 0;
  float* qu    = ws + off; off += (size_t)ROWS_ * MP_;   // FAVOR u / q features (in place)
  float* ku    = ws + off; off += (size_t)ROWS_ * MP_;
  float* red   = ws + off; off += 1024;                  // [0]=kmax, [1..]=partials
  float* perf  = ws + off; off += (size_t)ROWS_ * 64;
  float* pref  = ws + off; off += (size_t)H_ * (T_ + 1) * 64;
  float* h1    = ws + off; off += (size_t)ROWS_ * 128;
  float* dec   = ws + off; off += (size_t)ROWS_ * 128;
  float* c0    = ws + off; off += (size_t)48 * 1024 * 64;
  float* c1    = ws + off; off += (size_t)48 * 1024 * 64;
  float* c2    = ws + off; off += (size_t)48 * 1024 * 64;
  float* c2o   = ws + off; off += (size_t)ROWS_ * 128;
  float* probs = ws + off; off += (size_t)ROWS_ * 128;

  // 1) FAVOR feature GEMMs (WMMA bf16)
  favor_u_gemm<<<3264, 256, 0, stream>>>(q, proj, qu);
  favor_u_gemm<<<3264, 256, 0, stream>>>(k, proj, ku);
  // 2) key stabilizer = global max
  ku_rowmax_partial<<<96, 256, 0, stream>>>(ku, red + 1);
  ku_max_final<<<1, 128, 0, stream>>>(red + 1, 96, red);
  // 3) exp/feature map
  favor_exp<<<3072, 256, 0, stream>>>(qu, q, nullptr, 1);
  favor_exp<<<3072, 256, 0, stream>>>(ku, k, red, 0);
  // 4) causal linear attention (LDS-resident state)
  size_t attn_lds = (size_t)(M_ * 64 + MP_ * 3 + 64 + 256) * sizeof(float);
  causal_attn<<<H_, 256, attn_lds, stream>>>(qu, ku, v, perf);
  // 5-6) predictor MLP (WMMA bf16, TDM-staged weights)
  size_t enc_lds = (size_t)(192 * 128 + 16 * 132 + 32) * sizeof(float);
  encoder_gemm<<<1536, 256, enc_lds, stream>>>(perf, v, w_enc, b_enc, ln_g, ln_b, h1);
  decoder_gemm<<<1536, 256, 128 * 128 * sizeof(float), stream>>>(h1, w_dec, b_dec, dec);
  // 7) CNN via implicit-GEMM WMMA (TDM-staged filters)
  size_t w108 = (size_t)48 * 108 * sizeof(float);
  size_t w432 = (size_t)48 * 432 * sizeof(float);
  size_t w12  = (size_t)12 * 432 * sizeof(float);
  //    conv1: [12,2048,128] -> [48,1024,64], stride 2, relu
  conv3x3_wmma<<<1536, 256, w108, stream>>>(dec, c1_w, c1_b, nullptr, c0,
      12, 48, 2048, 128, 1024, 6, 2, 2, 0, 1, 3, 12288);
  //    resblock 1
  conv3x3_wmma<<<1536, 256, w432, stream>>>(c0, r1w1, r1b1, nullptr, c1,
      48, 48, 1024, 64, 1024, 6, 1, 1, 0, 1, 3, 12288);
  conv3x3_wmma<<<1536, 256, w432, stream>>>(c1, r1w2, r1b2, c0, c2,
      48, 48, 1024, 64, 1024, 6, 1, 1, 0, 1, 3, 12288);
  //    resblock 2
  conv3x3_wmma<<<1536, 256, w432, stream>>>(c2, r2w1, r2b1, nullptr, c0,
      48, 48, 1024, 64, 1024, 6, 1, 1, 0, 1, 3, 12288);
  conv3x3_wmma<<<1536, 256, w432, stream>>>(c0, r2w2, r2b2, c2, c1,
      48, 48, 1024, 64, 1024, 6, 1, 1, 0, 1, 3, 12288);
  //    c2 conv over nearest-2x upsampled input: -> [12,2048,128]
  conv3x3_wmma<<<2048, 256, w12, stream>>>(c1, c2_w, c2_b, nullptr, c2o,
      48, 12, 1024, 64, 2048, 7, 1, 1, 1, 0, 1, 16384);
  // 8) scores + softmax
  score_softmax<<<ROWS_, 128, 0, stream>>>(dec, c2o, probs);
  // 9-10) bucketed grid-sample attention via prefix sums (O(T*L*D))
  prefix_v<<<3, 256, 0, stream>>>(v, pref);
  sparse_interp_out<<<ROWS_, 128, 0, stream>>>(probs, pref, out);

  (void)in_sizes; (void)n_in; (void)out_size; (void)ws_size; (void)off;
}
